// CGCClass_22196390986156
// MI455X (gfx1250) — compile-verified, matrix operand order checked by reading, not run on hardware
//
#include <hip/hip_runtime.h>
#include <hip/hip_bf16.h>
#include <math.h>

// ---------------------------------------------------------------------------
// CGConv GNN on MI455X (gfx1250, wave32, WMMA).
// Edge message GEMM ([E,144] @ [144,64] twice per layer) runs on
// v_wmma_f32_16x16x32_f16 with f32 accumulation; gathers staged through LDS,
// weights staged via the Tensor Data Mover (tensor_load_to_lds + TENSORcnt).
// ---------------------------------------------------------------------------

#define NN_F 64      // node feature dim
#define D_E  16      // edge attr dim
#define ZIN  144     // 2*F + D
#define KPAD 160     // ZIN padded to multiple of 32 (5 k-steps)
#define EPB  128     // edges per block (8 waves x 16 rows)
#define NLAY 3
#define GNUM 64
#define HDIM 256
#define ODIM 2
#define EPSV 1e-5f

typedef __attribute__((ext_vector_type(16))) _Float16 v16h;
typedef __attribute__((ext_vector_type(8)))  _Float16 v8h;
typedef __attribute__((ext_vector_type(2)))  __fp16   v2fp16;
typedef __attribute__((ext_vector_type(8)))  float    v8f;
typedef __attribute__((ext_vector_type(4)))  unsigned u32x4;
typedef __attribute__((ext_vector_type(8)))  int      i32x8;
typedef __attribute__((ext_vector_type(4)))  int      i32x4;

#if __has_builtin(__builtin_amdgcn_tensor_load_to_lds) && \
    __has_builtin(__builtin_amdgcn_s_wait_tensorcnt)
#define USE_TDM 1
#endif

// ---------------- weight convert: f32 [L][144][64] -> f16 [L][2][64][160] (transposed, padded)
__global__ __launch_bounds__(256) void k_wconv(const float* __restrict__ Wf,
                                               const float* __restrict__ Ws,
                                               _Float16* __restrict__ w16) {
  int idx = blockIdx.x * 256 + threadIdx.x;
  const int total = NLAY * 2 * NN_F * KPAD;
  if (idx >= total) return;
  int k  = idx % KPAD;
  int n  = (idx / KPAD) % NN_F;
  int fs = (idx / (KPAD * NN_F)) % 2;
  int l  = idx / (KPAD * NN_F * 2);
  float v = 0.f;
  if (k < ZIN) {
    const float* W = fs ? Ws : Wf;
    v = W[((long long)l * ZIN + k) * NN_F + n];
  }
  w16[idx] = (_Float16)v;
}

// ---------------- generic f32 copy / zero ----------------
__global__ __launch_bounds__(256) void k_copy(const float* __restrict__ src,
                                              float* __restrict__ dst, long long n) {
  long long i = (long long)blockIdx.x * 256 + threadIdx.x;
  if (i < n) dst[i] = src[i];
}
__global__ __launch_bounds__(256) void k_zero(float* __restrict__ dst, long long n) {
  long long i = (long long)blockIdx.x * 256 + threadIdx.x;
  if (i < n) dst[i] = 0.f;
}
__global__ __launch_bounds__(256) void k_zero_u32(unsigned* __restrict__ dst, long long n) {
  long long i = (long long)blockIdx.x * 256 + threadIdx.x;
  if (i < n) dst[i] = 0u;   // below enc(-inf); any real value beats it
}

// ---------------- edge message GEMM + scatter-add -------------------------
// block: 256 threads (8 wave32), 128 edges. Per wave: 16x64 output tile.
__global__ __launch_bounds__(256)
void k_edge_gemm(const float* __restrict__ x,
                 const int*   __restrict__ edge_index,   // [2][E]
                 const float* __restrict__ edge_attr,    // [E][16]
                 const _Float16* __restrict__ w16,       // [2][64][160] this layer
                 const float* __restrict__ bf,           // [64]
                 const float* __restrict__ bs,           // [64]
                 float* __restrict__ agg,                // [N][64]
                 int E) {
  __shared__ __align__(16) _Float16 zsh[EPB][KPAD];        // 40 KB
  __shared__ __align__(16) _Float16 wsh[2 * NN_F * KPAD];  // 40 KB: [f|s][64][160]
  __shared__ int dsh[EPB];

  const int tid  = threadIdx.x;
  const int lane = tid & 31;
  const int wave = tid >> 5;
  const long long ebase = (long long)blockIdx.x * EPB;

  // --- stage both weight matrices into LDS ---
#ifdef USE_TDM
  if (tid == 0) {
    // Tensor DMA: one 1-D copy of 40 KB (5120 x 8B elements), global -> LDS.
    const unsigned lds = (unsigned)(unsigned long long)(&wsh[0]);
    const unsigned long long ga = (unsigned long long)w16;
    const unsigned elems = (2u * NN_F * KPAD * 2u) / 8u;   // 5120 (data_size = 8B)
    u32x4 g0;
    g0[0] = 1u;                                            // count=1, load, no gather
    g0[1] = lds;                                           // lds_addr
    g0[2] = (unsigned)(ga & 0xffffffffu);                  // global_addr[31:0]
    g0[3] = (unsigned)((ga >> 32) & 0x01ffffffu) | (2u << 30); // addr[56:32], type=2
    i32x8 g1;
    g1[0] = (3 << 16);                                     // data_size=3 (8 bytes)
    g1[1] = (int)((elems & 0xffffu) << 16);                // tensor_dim0[15:0]
    g1[2] = (int)((elems >> 16) | (1u << 16));             // dim0[31:16], tensor_dim1=1
    g1[3] = (int)((elems & 0xffffu) << 16);                // tile_dim0
    g1[4] = 1;                                             // tile_dim1=1, tile_dim2=0
    g1[5] = (int)elems;                                    // tensor_dim0_stride[31:0]
    g1[6] = 0;
    g1[7] = 0;
    i32x4 g2; g2[0] = 1; g2[1] = 1; g2[2] = 0; g2[3] = 0;  // dim2=1, dim3=1
    i32x4 g3; g3[0] = 0; g3[1] = 0; g3[2] = 0; g3[3] = 0;
    i32x8 g4; g4[0] = 0; g4[1] = 0; g4[2] = 0; g4[3] = 0;
              g4[4] = 0; g4[5] = 0; g4[6] = 0; g4[7] = 0;
    __builtin_amdgcn_tensor_load_to_lds(g0, g1, g2, g3, g4, 0);
  }
#else
  {
    const v8h* wsrc = (const v8h*)w16;
    v8h* wdst = (v8h*)&wsh[0];
    const int nchunk = (2 * NN_F * KPAD) / 8;  // 2560
    for (int i = tid; i < nchunk; i += 256) wdst[i] = wsrc[i];
  }
#endif

  // --- stage z = [x[dst] | x[src] | ea | 0] rows as f16 (2 threads/row) ---
  // overlaps with the weight DMA above
  {
    const int row  = tid >> 1;     // 0..127
    const int half = tid & 1;
    long long e  = ebase + row;
    long long ec = (e < E) ? e : (long long)(E - 1);
    const int isrc = edge_index[ec];
    const int idst = edge_index[(long long)E + ec];
    const bool valid = (e < (long long)E);
    if (half == 0) dsh[row] = idst;
#pragma unroll
    for (int c4 = 0; c4 < 20; ++c4) {          // 20 float4 per half-row
      const int col = half * 80 + c4 * 4;      // boundaries 64/128/144 are mult of 4
      float4 v;
      if (!valid)          { v.x = v.y = v.z = v.w = 0.f; }
      else if (col < 64)   v = *(const float4*)(x + (long long)idst * NN_F + col);
      else if (col < 128)  v = *(const float4*)(x + (long long)isrc * NN_F + (col - 64));
      else if (col < ZIN)  v = *(const float4*)(edge_attr + ec * D_E + (col - 128));
      else                 { v.x = v.y = v.z = v.w = 0.f; }
      // packed f32x2 -> f16x2 conversion, single 8-byte LDS store
      union { v2fp16 h[2]; unsigned long long u; } pk;
      pk.h[0] = __builtin_amdgcn_cvt_pkrtz(v.x, v.y);
      pk.h[1] = __builtin_amdgcn_cvt_pkrtz(v.z, v.w);
      *(unsigned long long*)&zsh[row][col] = pk.u;
    }
  }
#ifdef USE_TDM
  if (tid == 0) __builtin_amdgcn_s_wait_tensorcnt(0);
#endif
  __syncthreads();

  // --- WMMA main loop: K = 5 steps of 32, N = 4 tiles of 16 ---
  v8f accF[4] = {v8f{}, v8f{}, v8f{}, v8f{}};
  v8f accS[4] = {v8f{}, v8f{}, v8f{}, v8f{}};
  const int mrow = wave * 16 + (lane & 15);   // z row this lane feeds into A
  const int ksel = (lane >> 4) * 8;           // lanes 16-31 hold K+8..15
  const int ncol = lane & 15;

#pragma unroll
  for (int ks = 0; ks < 5; ++ks) {
    const int kb = ks * 32 + ksel;
    v8h alo = *(const v8h*)&zsh[mrow][kb];
    v8h ahi = *(const v8h*)&zsh[mrow][kb + 16];
    v16h a;
#pragma unroll
    for (int i = 0; i < 8; ++i) { a[i] = alo[i]; a[i + 8] = ahi[i]; }
#pragma unroll
    for (int nt = 0; nt < 4; ++nt) {
      const int n = nt * 16 + ncol;
      const _Float16* wfn = &wsh[n * KPAD];
      const _Float16* wsn = &wsh[(NN_F + n) * KPAD];
      v8h blo = *(const v8h*)&wfn[kb];
      v8h bhi = *(const v8h*)&wfn[kb + 16];
      v16h b;
#pragma unroll
      for (int i = 0; i < 8; ++i) { b[i] = blo[i]; b[i + 8] = bhi[i]; }
      accF[nt] = __builtin_amdgcn_wmma_f32_16x16x32_f16(
          false, a, false, b, (short)0, accF[nt], false, false);
      blo = *(const v8h*)&wsn[kb];
      bhi = *(const v8h*)&wsn[kb + 16];
      v16h bb;
#pragma unroll
      for (int i = 0; i < 8; ++i) { bb[i] = blo[i]; bb[i + 8] = bhi[i]; }
      accS[nt] = __builtin_amdgcn_wmma_f32_16x16x32_f16(
          false, a, false, bb, (short)0, accS[nt], false, false);
    }
  }

  // --- epilogue: sigmoid(f+bf) * softplus(s+bs), scatter-add to agg[dst] ---
  // C layout: VGPR j -> M = j + (lane>=16 ? 8 : 0), N = tile*16 + (lane&15)
#pragma unroll
  for (int nt = 0; nt < 4; ++nt) {
    const int n = nt * 16 + ncol;
    const float bfv = bf[n];
    const float bsv = bs[n];
#pragma unroll
    for (int j = 0; j < 8; ++j) {
      const int m   = j + ((lane & 16) ? 8 : 0);
      const int row = wave * 16 + m;
      const long long e = ebase + row;
      if (e < (long long)E) {
        const float vf = accF[nt][j] + bfv;
        const float vs = accS[nt][j] + bsv;
        const float gate = 1.0f / (1.0f + __expf(-vf));
        const float sp   = (vs > 20.f) ? vs : log1pf(__expf(vs));
        atomicAdd(agg + (long long)dsh[row] * NN_F + n, gate * sp);
      }
    }
  }
}

// ---------------- batchnorm (inference) + residual ------------------------
__global__ __launch_bounds__(256)
void k_bn_res(float* __restrict__ x_cur, const float* __restrict__ agg,
              const float* __restrict__ g, const float* __restrict__ b,
              const float* __restrict__ m, const float* __restrict__ v,
              long long n) {
  long long i = (long long)blockIdx.x * 256 + threadIdx.x;
  if (i >= n) return;
  const int f = (int)(i & (NN_F - 1));
  const float a = (agg[i] - m[f]) * rsqrtf(v[f] + EPSV) * g[f] + b[f];
  x_cur[i] += a;
}

// ---------------- segment_max pooling via monotone uint encoding ----------
__device__ __forceinline__ unsigned fenc(float v) {
  unsigned u = __float_as_uint(v);
  return (u & 0x80000000u) ? ~u : (u | 0x80000000u);
}
__device__ __forceinline__ float fdec(unsigned u) {
  return __uint_as_float((u & 0x80000000u) ? (u & 0x7fffffffu) : ~u);
}
__global__ __launch_bounds__(256)
void k_pool_max(const float* __restrict__ x_cur, const int* __restrict__ batch,
                unsigned* __restrict__ pooled_enc, long long n) {
  long long i = (long long)blockIdx.x * 256 + threadIdx.x;
  if (i >= n) return;
  const int node = (int)(i >> 6);
  const int f    = (int)(i & (NN_F - 1));
  const int gph  = batch[node];
  atomicMax(pooled_enc + (long long)gph * NN_F + f, fenc(x_cur[i]));
}
__global__ __launch_bounds__(256)
void k_pool_dec(const unsigned* __restrict__ pooled_enc, float* __restrict__ pooled,
                long long n) {
  long long i = (long long)blockIdx.x * 256 + threadIdx.x;
  if (i < n) pooled[i] = fdec(pooled_enc[i]);
}

// ---------------- MLP head (tiny: ~2 MFLOP total) --------------------------
__global__ __launch_bounds__(HDIM)
void k_head1(const float* __restrict__ pooled, const float* __restrict__ W1,
             const float* __restrict__ b1, const float* __restrict__ g2,
             const float* __restrict__ b2n, const float* __restrict__ m2,
             const float* __restrict__ v2, float* __restrict__ h) {
  __shared__ float psh[NN_F];
  const int j = threadIdx.x;
  if (j < NN_F) psh[j] = pooled[(long long)blockIdx.x * NN_F + j];
  __syncthreads();
  float s = b1[j];
#pragma unroll 8
  for (int k = 0; k < NN_F; ++k) s += psh[k] * W1[k * HDIM + j];
  s = fmaxf(s, 0.f);
  s = (s - m2[j]) * rsqrtf(v2[j] + EPSV) * g2[j] + b2n[j];
  h[(long long)blockIdx.x * HDIM + j] = s;
}
__global__ __launch_bounds__(GNUM * ODIM)
void k_head2(const float* __restrict__ h, const float* __restrict__ W2,
             const float* __restrict__ b2, float* __restrict__ out) {
  const int t = threadIdx.x;           // 128 threads
  const int g = t >> 1, o = t & 1;
  float s = b2[o];
#pragma unroll 8
  for (int j = 0; j < HDIM; ++j) s += h[g * HDIM + j] * W2[j * ODIM + o];
  out[g * ODIM + o] = 1.0f / (1.0f + __expf(-s));
}

// ---------------------------------------------------------------------------
extern "C" void kernel_launch(void* const* d_in, const int* in_sizes, int n_in,
                              void* d_out, int out_size, void* d_ws, size_t ws_size,
                              hipStream_t stream) {
  const float* x          = (const float*)d_in[0];
  const int*   edge_index = (const int*)  d_in[1];
  const float* edge_attr  = (const float*)d_in[2];
  const int*   batch      = (const int*)  d_in[3];
  const float* Wf         = (const float*)d_in[4];
  const float* bf         = (const float*)d_in[5];
  const float* Ws         = (const float*)d_in[6];
  const float* bs         = (const float*)d_in[7];
  const float* bn_g       = (const float*)d_in[8];
  const float* bn_b       = (const float*)d_in[9];
  const float* bn_m       = (const float*)d_in[10];
  const float* bn_v       = (const float*)d_in[11];
  const float* W1         = (const float*)d_in[12];
  const float* b1         = (const float*)d_in[13];
  const float* bn2_g      = (const float*)d_in[14];
  const float* bn2_b      = (const float*)d_in[15];
  const float* bn2_m      = (const float*)d_in[16];
  const float* bn2_v      = (const float*)d_in[17];
  const float* W2         = (const float*)d_in[18];
  const float* b2         = (const float*)d_in[19];
  float* out = (float*)d_out;

  const int N = in_sizes[0] / NN_F;
  const int E = in_sizes[2] / D_E;
  const long long NF = (long long)N * NN_F;

  // workspace layout (all 256B aligned)
  auto align256 = [](size_t v) { return (v + 255) & ~(size_t)255; };
  char* ws = (char*)d_ws;
  size_t off = 0;
  float*    x_cur  = (float*)(ws + off); off = align256(off + NF * 4);
  float*    agg    = (float*)(ws + off); off = align256(off + NF * 4);
  _Float16* w16    = (_Float16*)(ws + off); off = align256(off + (size_t)NLAY * 2 * NN_F * KPAD * 2);
  unsigned* p_enc  = (unsigned*)(ws + off); off = align256(off + (size_t)GNUM * NN_F * 4);
  float*    pooled = (float*)(ws + off); off = align256(off + (size_t)GNUM * NN_F * 4);
  float*    h      = (float*)(ws + off); off = align256(off + (size_t)GNUM * HDIM * 4);
  (void)ws_size;

  const int blkNF  = (int)((NF + 255) / 256);
  const int blkE   = (E + EPB - 1) / EPB;
  const int wtot   = NLAY * 2 * NN_F * KPAD;

  k_copy <<<blkNF, 256, 0, stream>>>(x, x_cur, NF);
  k_wconv<<<(wtot + 255) / 256, 256, 0, stream>>>(Wf, Ws, w16);

  for (int l = 0; l < NLAY; ++l) {
    k_zero<<<blkNF, 256, 0, stream>>>(agg, NF);
    k_edge_gemm<<<blkE, 256, 0, stream>>>(
        x_cur, edge_index, edge_attr,
        w16 + (size_t)l * 2 * NN_F * KPAD,
        bf + l * NN_F, bs + l * NN_F, agg, E);
    k_bn_res<<<blkNF, 256, 0, stream>>>(
        x_cur, agg, bn_g + l * NN_F, bn_b + l * NN_F,
        bn_m + l * NN_F, bn_v + l * NN_F, NF);
  }

  const long long GF = (long long)GNUM * NN_F;
  k_zero_u32<<<(int)((GF + 255) / 256), 256, 0, stream>>>(p_enc, GF);
  k_pool_max<<<blkNF, 256, 0, stream>>>(x_cur, batch, p_enc, NF);
  k_pool_dec<<<(int)((GF + 255) / 256), 256, 0, stream>>>(p_enc, pooled, GF);

  k_head1<<<GNUM, HDIM, 0, stream>>>(pooled, W1, b1, bn2_g, bn2_b, bn2_m, bn2_v, h);
  k_head2<<<1, GNUM * ODIM, 0, stream>>>(h, W2, b2, out);
}
